// MACModuleWithGradientUpdate_35948876267556
// MI455X (gfx1250) — matrix operand order repaired and checked by reading, not compiled, not run on hardware
//
#include <hip/hip_runtime.h>
#include <math.h>

#define DIM    1024
#define BATCH  32
#define TSTEPS 2048
#define MROWS  131072
#define TOPK   8

typedef __attribute__((ext_vector_type(16))) __bf16        v16bf;
typedef __attribute__((ext_vector_type(8)))  float         v8f;
typedef __attribute__((ext_vector_type(8)))  unsigned int  v8u;

__device__ __forceinline__ unsigned short f32_to_bf16_rne(float f) {
    unsigned int u = __float_as_uint(f);
    unsigned int r = u + 0x7FFFu + ((u >> 16) & 1u);
    return (unsigned short)(r >> 16);
}
__device__ __forceinline__ float bf16_to_f32(unsigned short h) {
    return __uint_as_float(((unsigned int)h) << 16);
}

// ---------------------------------------------------------------------------
// Kernel 1: partial mean over time. grid (4 cchunks, 8 tchunks, 32 batch)
// ---------------------------------------------------------------------------
__global__ void k_mean_partial(const float* __restrict__ se, float* __restrict__ partial) {
    int c  = blockIdx.x * 256 + threadIdx.x;   // 0..1023
    int tc = blockIdx.y;                       // 0..7
    int b  = blockIdx.z;                       // 0..31
    const float* p = se + ((size_t)b * TSTEPS + (size_t)tc * 256) * DIM + c;
    float acc = 0.0f;
    #pragma unroll 8
    for (int i = 0; i < 256; ++i) acc += p[(size_t)i * DIM];
    partial[((size_t)b * 8 + tc) * DIM + c] = acc;
}

// ---------------------------------------------------------------------------
// Kernel 2: finalize mean. 32768 outputs.
// ---------------------------------------------------------------------------
__global__ void k_mean_final(const float* __restrict__ partial, float* __restrict__ meanp) {
    int g = blockIdx.x * 256 + threadIdx.x;    // 0..32767
    int b = g >> 10, c = g & 1023;
    float acc = 0.0f;
    #pragma unroll
    for (int j = 0; j < 8; ++j) acc += partial[((size_t)b * 8 + j) * DIM + c];
    meanp[g] = acc * (1.0f / (float)TSTEPS);
}

// ---------------------------------------------------------------------------
// Kernel 3: q = (mean @ Wq^T + bq) / 32, split into bf16 hi/lo (truncation
// split: hi = upper 16 bits, lo = rne(q - hi)).
// One wave per output element. 32768 waves -> 4096 blocks of 256.
// ---------------------------------------------------------------------------
__global__ void k_qproj(const float* __restrict__ meanp, const float* __restrict__ Wq,
                        const float* __restrict__ bq,
                        unsigned short* __restrict__ qhi, unsigned short* __restrict__ qlo) {
    int w    = (blockIdx.x * 256 + threadIdx.x) >> 5;  // global wave / output id
    int lane = threadIdx.x & 31;
    int b = w >> 10, i = w & 1023;
    const float* mp = meanp + (size_t)b * DIM;
    const float* wp = Wq + (size_t)i * DIM;
    float acc = 0.0f;
    #pragma unroll 8
    for (int j = 0; j < 32; ++j) {
        int c = j * 32 + lane;
        acc += mp[c] * wp[c];
    }
    #pragma unroll
    for (int off = 16; off > 0; off >>= 1) acc += __shfl_xor(acc, off, 32);
    if (lane == 0) {
        float q = (acc + bq[i]) * (1.0f / 32.0f);   // fold 1/sqrt(1024)
        unsigned short h = (unsigned short)(__float_as_uint(q) >> 16);  // truncate
        float lo = q - bf16_to_f32(h);
        qhi[w] = h;
        qlo[w] = f32_to_bf16_rne(lo);
    }
}

// ---------------------------------------------------------------------------
// Kernel 4: scores(32 x 131072) = (q/32) @ mem_bank^T via bf16 split WMMA.
// One wave per 16-row N-tile; 8192 tiles; 8 waves/block -> 1024 blocks.
// Lanes (L, L+16) share a mem row and consume alternating 32B quarters, so each
// 128B cacheline is fully consumed within one k-chunk.
// Truncation split: hi = x & 0xffff0000 (packed via v_perm_b32), lo = x - hi
// (exact); dropped terms ~2^-16 relative -- far below top-k score gaps.
// ---------------------------------------------------------------------------
__global__ void __launch_bounds__(256) k_scores(const unsigned short* __restrict__ qhi,
                                                const unsigned short* __restrict__ qlo,
                                                const float* __restrict__ mem,
                                                float* __restrict__ scores) {
    const int lane  = threadIdx.x & 31;
    const int tile  = blockIdx.x * 8 + (threadIdx.x >> 5);
    const int nbase = tile * 16;
    const int nlo   = lane & 15;
    const int hiL   = lane >> 4;

    const float* rowp = mem + (size_t)(nbase + nlo) * DIM;
    const unsigned int* qh32 = (const unsigned int*)qhi;
    const unsigned int* ql32 = (const unsigned int*)qlo;

    // A-layout per-lane u32 (bf16 pair) offsets within a 32-K chunk:
    // raw k offsets {0,2,4,6,16,18,20,22} + hiL*8  ->  /2 = {0,1,2,3,8,9,10,11} + hiL*4
    const int m0 = nlo;        // M rows 0..15  (batch 0..15)
    const int m1 = nlo + 16;   // batch 16..31
    const int a0base = m0 * (DIM / 2) + hiL * 4;
    const int a1base = m1 * (DIM / 2) + hiL * 4;

    v8f c0 = {};
    v8f c1 = {};

    for (int kc = 0; kc < DIM / 32; ++kc) {
        const int kb = kc * 32;
        // ---- B tile: fp32 k offsets {0..7,16..23} + hiL*8 of this lane's mem row ----
        const float4* fp = (const float4*)(rowp + kb + hiL * 8);
        float f[16];
        #pragma unroll
        for (int q4 = 0; q4 < 4; ++q4) {
            float4 x = fp[(q4 & 1) + (q4 >> 1) * 4];
            f[q4 * 4 + 0] = x.x; f[q4 * 4 + 1] = x.y;
            f[q4 * 4 + 2] = x.z; f[q4 * 4 + 3] = x.w;
        }
        v8u bh, bl;
        #pragma unroll
        for (int j = 0; j < 8; ++j) {
            float x0 = f[2 * j], x1 = f[2 * j + 1];
            unsigned int u0 = __float_as_uint(x0);
            unsigned int u1 = __float_as_uint(x1);
            // packed truncated-bf16 pair: [15:0]=x0_hi, [31:16]=x1_hi
            bh[j] = __builtin_amdgcn_perm(u1, u0, 0x07060302u);
            float h0 = __uint_as_float(u0 & 0xFFFF0000u);
            float h1 = __uint_as_float(u1 & 0xFFFF0000u);
            float l0 = x0 - h0;   // exact residual
            float l1 = x1 - h1;
            bl[j] = __builtin_amdgcn_perm(__float_as_uint(l1), __float_as_uint(l0),
                                          0x07060302u);
        }
        v16bf Bh = __builtin_bit_cast(v16bf, bh);
        v16bf Bl = __builtin_bit_cast(v16bf, bl);

        // ---- A tiles (hi/lo, 2 M-tiles) ----
        const int cjs[8] = {0, 1, 2, 3, 8, 9, 10, 11};
        v8u a0h, a0l, a1h, a1l;
        #pragma unroll
        for (int j = 0; j < 8; ++j) {
            int i0 = a0base + kc * 16 + cjs[j];
            int i1 = a1base + kc * 16 + cjs[j];
            a0h[j] = qh32[i0]; a0l[j] = ql32[i0];
            a1h[j] = qh32[i1]; a1l[j] = ql32[i1];
        }
        v16bf A0h = __builtin_bit_cast(v16bf, a0h);
        v16bf A0l = __builtin_bit_cast(v16bf, a0l);
        v16bf A1h = __builtin_bit_cast(v16bf, a1h);
        v16bf A1l = __builtin_bit_cast(v16bf, a1l);

        // ---- split-precision accumulate: hh + hl + lh ----
        c0 = __builtin_amdgcn_wmma_f32_16x16x32_bf16(false, A0h, false, Bh, (short)0, c0, false, false);
        c0 = __builtin_amdgcn_wmma_f32_16x16x32_bf16(false, A0h, false, Bl, (short)0, c0, false, false);
        c0 = __builtin_amdgcn_wmma_f32_16x16x32_bf16(false, A0l, false, Bh, (short)0, c0, false, false);
        c1 = __builtin_amdgcn_wmma_f32_16x16x32_bf16(false, A1h, false, Bh, (short)0, c1, false, false);
        c1 = __builtin_amdgcn_wmma_f32_16x16x32_bf16(false, A1h, false, Bl, (short)0, c1, false, false);
        c1 = __builtin_amdgcn_wmma_f32_16x16x32_bf16(false, A1l, false, Bh, (short)0, c1, false, false);
    }

    // C/D layout: VGPR v, lanes 0-15 -> M=v, lanes 16-31 -> M=v+8; N=lane&15
    #pragma unroll
    for (int v = 0; v < 8; ++v) {
        int mA = v + 8 * hiL;
        scores[(size_t)mA * MROWS + nbase + nlo]        = c0[v];
        scores[(size_t)(mA + 16) * MROWS + nbase + nlo] = c1[v];
    }
}

// ---------------------------------------------------------------------------
// Kernel 5: per-batch top-8 -> softmax -> weighted gather. 1 block per batch.
// ---------------------------------------------------------------------------
__device__ __forceinline__ void topk_insert(float (&s)[8], int (&ix)[8], float v, int n) {
    if (v > s[7]) {
        s[7] = v; ix[7] = n;
        #pragma unroll
        for (int j = 7; j > 0; --j) {
            if (s[j] > s[j - 1]) {
                float ts = s[j]; s[j] = s[j - 1]; s[j - 1] = ts;
                int ti = ix[j]; ix[j] = ix[j - 1]; ix[j - 1] = ti;
            }
        }
    }
}

__global__ void __launch_bounds__(256) k_topk_gather(const float* __restrict__ scores,
                                                     const float* __restrict__ mem,
                                                     float* __restrict__ out) {
    __shared__ float cs[2048];
    __shared__ int   ci[2048];
    __shared__ float cs2[256];
    __shared__ int   ci2[256];
    __shared__ float fw[8];
    __shared__ int   fidx[8];

    const int b = blockIdx.x;
    const int t = threadIdx.x;
    const float* sr = scores + (size_t)b * MROWS;

    float s8[8]; int i8[8];
    #pragma unroll
    for (int j = 0; j < 8; ++j) { s8[j] = -3.0e38f; i8[j] = 0; }
    for (int n = t; n < MROWS; n += 256) topk_insert(s8, i8, sr[n], n);
    #pragma unroll
    for (int j = 0; j < 8; ++j) { cs[t * 8 + j] = s8[j]; ci[t * 8 + j] = i8[j]; }
    __syncthreads();

    if (t < 32) {
        #pragma unroll
        for (int j = 0; j < 8; ++j) { s8[j] = -3.0e38f; i8[j] = 0; }
        for (int n = t * 64; n < t * 64 + 64; ++n) topk_insert(s8, i8, cs[n], ci[n]);
        #pragma unroll
        for (int j = 0; j < 8; ++j) { cs2[t * 8 + j] = s8[j]; ci2[t * 8 + j] = i8[j]; }
    }
    __syncthreads();

    if (t == 0) {
        #pragma unroll
        for (int j = 0; j < 8; ++j) { s8[j] = -3.0e38f; i8[j] = 0; }
        for (int n = 0; n < 256; ++n) topk_insert(s8, i8, cs2[n], ci2[n]);
        float mx = s8[0];
        float wsum = 0.0f;
        float w[8];
        #pragma unroll
        for (int j = 0; j < 8; ++j) { w[j] = __expf(s8[j] - mx); wsum += w[j]; }
        float inv = 1.0f / wsum;
        #pragma unroll
        for (int j = 0; j < 8; ++j) { fw[j] = w[j] * inv; fidx[j] = i8[j]; }
    }
    __syncthreads();

    for (int c = t; c < DIM; c += 256) {
        float acc = 0.0f;
        #pragma unroll
        for (int j = 0; j < 8; ++j) acc += fw[j] * mem[(size_t)fidx[j] * DIM + c];
        out[(size_t)b * DIM + c] = acc;
    }
}

// ---------------------------------------------------------------------------
extern "C" void kernel_launch(void* const* d_in, const int* in_sizes, int n_in,
                              void* d_out, int out_size, void* d_ws, size_t ws_size,
                              hipStream_t stream) {
    const float* se  = (const float*)d_in[0];  // (32, 2048, 1024)
    const float* Wq  = (const float*)d_in[1];  // (1024, 1024)
    const float* bq  = (const float*)d_in[2];  // (1024,)
    const float* mem = (const float*)d_in[3];  // (131072, 1024)
    // d_in[4] = k (== 8, hardcoded)
    float* out = (float*)d_out;                // (32, 1, 1024)

    char* wsb = (char*)d_ws;
    float*          partial = (float*)(wsb);                       // 1,048,576 B
    float*          meanp   = (float*)(wsb + 1048576);             //   131,072 B
    unsigned short* qhi     = (unsigned short*)(wsb + 1179648);    //    65,536 B
    unsigned short* qlo     = (unsigned short*)(wsb + 1245184);    //    65,536 B
    float*          scores  = (float*)(wsb + 1310720);             // 16,777,216 B

    k_mean_partial<<<dim3(4, 8, 32), 256, 0, stream>>>(se, partial);
    k_mean_final<<<128, 256, 0, stream>>>(partial, meanp);
    k_qproj<<<4096, 256, 0, stream>>>(meanp, Wq, bq, qhi, qlo);
    k_scores<<<MROWS / 16 / 8, 256, 0, stream>>>(qhi, qlo, mem, scores);
    k_topk_gather<<<BATCH, 256, 0, stream>>>(scores, mem, out);
}
